// CharacterEmbeddingLayer_26139170964177
// MI455X (gfx1250) — matrix operand order repaired and checked by reading, not compiled, not run on hardware
//
#include <hip/hip_runtime.h>
#include <hip/hip_bf16.h>

typedef __attribute__((ext_vector_type(16))) __bf16       v16bf;
typedef __attribute__((ext_vector_type(8)))  float        v8f;
typedef __attribute__((ext_vector_type(4)))  unsigned int v4u;

union Frag { v4u seg[2]; v16bf v; };

// Workspace byte offsets
#define WS_CV    0          //  96*64 bf16                       = 12,288 B
#define WS_FILT  16384      //  conv filter fragments            = 200,704 B
#define WS_PROJ  217088     //  w_proj fragments (K padded 416)  = 106,496 B
#define WS_HW    323584     //  g_w0,t_w0,g_w1,t_w1 fragments    = 131,072 B
#define WS_OUTS  454656     //  pooled features bf16 [25600][416]= 21,299,200 B

#define TPW 4               // tokens per wave in conv kernel

__device__ __forceinline__ __bf16 f2bf(float f) {
  unsigned int u = __float_as_uint(f);
  u = (u + 0x7fffu + ((u >> 16) & 1u)) >> 16;   // round-to-nearest-even
  unsigned short s = (unsigned short)u;
  __bf16 b;
  __builtin_memcpy(&b, &s, 2);
  return b;
}

__device__ __forceinline__ v8f vzero8() {
  v8f z = {0.f, 0.f, 0.f, 0.f, 0.f, 0.f, 0.f, 0.f};
  return z;
}

// Branchless hw-transcendental tanh: 1 - 2/(e^{2x}+1); saturates to +/-1.
__device__ __forceinline__ float tanh_fast(float x) {
  float e = __builtin_amdgcn_exp2f(x * 2.8853900817779268f);   // 2*log2(e)
  return 1.0f - 2.0f * __builtin_amdgcn_rcpf(e + 1.0f);
}

// Branchless sigmoid: 1/(1+e^{-z}).
__device__ __forceinline__ float sigmoid_fast(float z) {
  float e = __builtin_amdgcn_exp2f(z * -1.4426950408889634f);  // -log2(e)
  return __builtin_amdgcn_rcpf(1.0f + e);
}

// ---------------------------------------------------------------------------
// Prep: convert all weights to bf16 in fragment-ready (lane-ordered) layout.
// B-fragment layout (per 16x16x32 step): 1024 B, lane l owns halves
// [l*16 .. l*16+15] = W[nt*16 + l%16][ks*32 + (l>=16?16:0) + 0..15]
// ---------------------------------------------------------------------------
__global__ void prep_kernel(const float* cv, const float* f2, const float* f3,
                            const float* f4, const float* f5, const float* wproj,
                            const float* gw0, const float* tw0,
                            const float* gw1, const float* tw1, char* ws) {
  int gid = blockIdx.x * blockDim.x + threadIdx.x;
  __bf16 zero = f2bf(0.0f);

  if (gid < 6144) {                                    // char_vectors -> bf16
    ((__bf16*)(ws + WS_CV))[gid] = f2bf(cv[gid]);
    return;
  }
  gid -= 6144;
  if (gid < 100352) {                                  // conv filters
    int w, base; const float* fp;
    if (gid < 14336)      { w = 2; base = 0;     fp = f2; }
    else if (gid < 35840) { w = 3; base = 14336; fp = f3; }
    else if (gid < 64512) { w = 4; base = 35840; fp = f4; }
    else                  { w = 5; base = 64512; fp = f5; }
    int rel  = gid - base;
    int frag = rel >> 9;                 // 512 halves per fragment
    int nt   = frag / (2 * w);
    int ks   = frag % (2 * w);
    int inl  = rel & 511;
    int l    = inl >> 4, h = inl & 15;
    int f    = nt * 16 + (l & 15);
    int k    = ks * 32 + ((l >= 16) ? 16 : 0) + h;
    ((__bf16*)(ws + WS_FILT))[gid] = (f < 100) ? f2bf(fp[f * (w * 64) + k]) : zero;
    return;
  }
  gid -= 100352;
  if (gid < 53248) {                                   // w_proj (K 400 -> 416)
    int frag = gid >> 9;
    int nt = frag / 13, ks = frag % 13;
    int inl = gid & 511;
    int l = inl >> 4, h = inl & 15;
    int f = nt * 16 + (l & 15);
    int k = ks * 32 + ((l >= 16) ? 16 : 0) + h;
    ((__bf16*)(ws + WS_PROJ))[gid] = (k < 400) ? f2bf(wproj[f * 400 + k]) : zero;
    return;
  }
  gid -= 53248;
  if (gid < 65536) {                                   // highway weights
    int m = gid >> 14;                                 // 0=g_w0 1=t_w0 2=g_w1 3=t_w1
    const float* src = (m == 0) ? gw0 : (m == 1) ? tw0 : (m == 2) ? gw1 : tw1;
    int rel = gid & 16383;
    int frag = rel >> 9;
    int nt = frag >> 2, ks = frag & 3;
    int inl = rel & 511;
    int l = inl >> 4, h = inl & 15;
    int f = nt * 16 + (l & 15);
    int k = ks * 32 + ((l >= 16) ? 16 : 0) + h;
    ((__bf16*)(ws + WS_HW))[gid] = f2bf(src[f * 128 + k]);
    return;
  }
  gid -= 65536;
  if (gid < 409600) {                                  // zero outs pad cols 400..415
    int tok = gid >> 4, col = 400 + (gid & 15);
    ((__bf16*)(ws + WS_OUTS))[(size_t)tok * 416 + col] = zero;
  }
}

// ---------------------------------------------------------------------------
// Kernel 1: conv branches + tanh + max-pool. TPW tokens per wave; filter
// B-fragments held in VGPRs across the token loop (4x B-traffic reuse).
// A matrix for window w = shifted views of the token's 16x64 embedding (LDS).
// ---------------------------------------------------------------------------
__global__ __launch_bounds__(256) void conv_kernel(const int* __restrict__ char_idxs,
                                                   char* __restrict__ ws) {
  __shared__ __bf16 smE[8][TPW][1280];                 // 20 rows x 64 cols per token
  int lane = threadIdx.x & 31, wv = threadIdx.x >> 5;
  int t0 = (blockIdx.x * 8 + wv) * TPW;                // first token, grid = 800
  const __bf16* cvb = (const __bf16*)(ws + WS_CV);

  #pragma unroll
  for (int tk = 0; tk < TPW; ++tk) {                   // gather embedding tiles
    int t   = t0 + tk;
    int row = lane & 15;
    int ch  = (lane >> 4) * 32;                        // half-col 0 or 32
    int ci  = char_idxs[t * 16 + row];
    const v4u* src = (const v4u*)(cvb + ci * 64 + ch);
    v4u* dst = (v4u*)(&smE[wv][tk][row * 64 + ch]);
    dst[0] = src[0]; dst[1] = src[1]; dst[2] = src[2]; dst[3] = src[3];
    v4u z = {0u, 0u, 0u, 0u};                          // zero pad rows 16..19
    ((v4u*)&smE[wv][tk][1024])[lane] = z;
  }
  __syncthreads();

  const char* filtReg = ws + WS_FILT;
  int mb = (lane >> 4) << 3;                           // M-half offset (0/8); K-half offset is 2*mb (0/16)
  const int winBase[4] = {0, 14336, 35840, 64512};

  #pragma unroll
  for (int w = 2; w <= 5; ++w) {
    int L = 17 - w;                                    // valid positions
    int wbase = winBase[w - 2];
    for (int nt = 0; nt < 7; ++nt) {
      Frag bf_[10];                                    // B fragments stay resident
      #pragma unroll
      for (int ks = 0; ks < 2 * w; ++ks) {
        const v4u* bp = (const v4u*)(filtReg +
            (size_t)(wbase + (nt * 2 * w + ks) * 512) * 2 + lane * 32);
        bf_[ks].seg[0] = bp[0]; bf_[ks].seg[1] = bp[1];
      }
      #pragma unroll
      for (int tk = 0; tk < TPW; ++tk) {
        const __bf16* E = &smE[wv][tk][0];
        v8f acc = vzero8();
        #pragma unroll
        for (int ks = 0; ks < 2 * w; ++ks) {
          Frag a;
          int row = (lane & 15) + (ks >> 1);           // shifted embedding row
          int ch  = ((ks & 1) << 5) + 2 * mb;          // K-chunk base within row
          a.seg[0] = *(const v4u*)(E + row * 64 + ch);
          a.seg[1] = *(const v4u*)(E + row * 64 + ch + 8);
          acc = __builtin_amdgcn_wmma_f32_16x16x32_bf16(false, a.v, false, bf_[ks].v,
                                                        (short)0, acc, false, false);
        }
        // tanh + masked max-pool over positions (M dim)
        float m = -1e30f;
        #pragma unroll
        for (int r = 0; r < 8; ++r) {
          float h = tanh_fast(acc[r]);
          if ((r + mb) < L) m = fmaxf(m, h);
        }
        m = fmaxf(m, __shfl_xor(m, 16, 32));           // merge M halves (wave32)
        int f = nt * 16 + (lane & 15);
        if (lane < 16 && f < 100) {
          __bf16* outp = (__bf16*)(ws + WS_OUTS) + (size_t)(t0 + tk) * 416;
          outp[(w - 2) * 100 + f] = f2bf(m);
        }
      }
    }
  }
}

// ---------------------------------------------------------------------------
// Kernel 2: projection (400->128) + two fused highway layers. One 16-token
// tile per wave; x kept in f32 C-layout regs, bf16 copy in LDS for A-side.
// ---------------------------------------------------------------------------
__global__ __launch_bounds__(256) void head_kernel(
    const char* __restrict__ ws, const float* __restrict__ tb0,
    const float* __restrict__ gb0, const float* __restrict__ tb1,
    const float* __restrict__ gb1, float* __restrict__ out) {
  __shared__ __bf16 smX[8][2][2048];                   // per-wave double buffer
  int lane = threadIdx.x & 31, wv = threadIdx.x >> 5;
  int tt = blockIdx.x * 8 + wv;                        // tile id, grid = 200
  int laneN = lane & 15, mb = (lane >> 4) << 3;
  const __bf16* outs = (const __bf16*)(ws + WS_OUTS);
  const char* projReg = ws + WS_PROJ;
  const char* hwReg   = ws + WS_HW;

  v8f X[8];
  #pragma unroll
  for (int nt = 0; nt < 8; ++nt) {                     // x = outs @ w_proj^T
    v8f acc = vzero8();
    #pragma unroll
    for (int ks = 0; ks < 13; ++ks) {
      Frag a, b;
      const __bf16* ap = outs + (size_t)(tt * 16 + laneN) * 416 + ks * 32 + 2 * mb;
      a.seg[0] = *(const v4u*)ap;
      a.seg[1] = *(const v4u*)(ap + 8);
      const v4u* bp = (const v4u*)(projReg + (nt * 13 + ks) * 1024 + lane * 32);
      b.seg[0] = bp[0]; b.seg[1] = bp[1];
      acc = __builtin_amdgcn_wmma_f32_16x16x32_bf16(false, a.v, false, b.v,
                                                    (short)0, acc, false, false);
    }
    X[nt] = acc;
    #pragma unroll
    for (int r = 0; r < 8; ++r)
      smX[wv][0][(r + mb) * 128 + nt * 16 + laneN] = f2bf(acc[r]);
  }
  __syncthreads();

  #pragma unroll
  for (int layer = 0; layer < 2; ++layer) {
    const float* gB = layer ? gb1 : gb0;
    const float* tB = layer ? tb1 : tb0;
    #pragma unroll
    for (int nt = 0; nt < 8; ++nt) {
      v8f ag = vzero8(), at = vzero8();
      #pragma unroll
      for (int ks = 0; ks < 4; ++ks) {
        Frag a, bg, bt;
        const __bf16* ap = &smX[wv][layer][laneN * 128 + ks * 32 + 2 * mb];
        a.seg[0] = *(const v4u*)ap;
        a.seg[1] = *(const v4u*)(ap + 8);
        const v4u* bgp = (const v4u*)(hwReg + (2 * layer) * 32768 +
                                      (nt * 4 + ks) * 1024 + lane * 32);
        const v4u* btp = (const v4u*)(hwReg + (2 * layer + 1) * 32768 +
                                      (nt * 4 + ks) * 1024 + lane * 32);
        bg.seg[0] = bgp[0]; bg.seg[1] = bgp[1];
        bt.seg[0] = btp[0]; bt.seg[1] = btp[1];
        ag = __builtin_amdgcn_wmma_f32_16x16x32_bf16(false, a.v, false, bg.v,
                                                     (short)0, ag, false, false);
        at = __builtin_amdgcn_wmma_f32_16x16x32_bf16(false, a.v, false, bt.v,
                                                     (short)0, at, false, false);
      }
      int n = nt * 16 + laneN;
      float gbv = gB[n], tbv = tB[n];
      #pragma unroll
      for (int r = 0; r < 8; ++r) {
        float g  = sigmoid_fast(ag[r] + gbv);
        float tv = fmaxf(at[r] + tbv, 0.0f);
        float xn = g * tv + (1.0f - g) * X[nt][r];
        X[nt][r] = xn;
        if (layer == 0) smX[wv][1][(r + mb) * 128 + n] = f2bf(xn);
      }
    }
    __syncthreads();
  }

  #pragma unroll
  for (int nt = 0; nt < 8; ++nt) {
    #pragma unroll
    for (int r = 0; r < 8; ++r)
      out[(size_t)(tt * 16 + mb + r) * 128 + nt * 16 + laneN] = X[nt][r];
  }
}

extern "C" void kernel_launch(void* const* d_in, const int* in_sizes, int n_in,
                              void* d_out, int out_size, void* d_ws, size_t ws_size,
                              hipStream_t stream) {
  const int*   char_idxs = (const int*)d_in[0];
  const float* cv    = (const float*)d_in[1];
  const float* f2    = (const float*)d_in[2];
  const float* f3    = (const float*)d_in[3];
  const float* f4    = (const float*)d_in[4];
  const float* f5    = (const float*)d_in[5];
  const float* wproj = (const float*)d_in[6];
  const float* tw0   = (const float*)d_in[7];
  const float* tb0   = (const float*)d_in[8];
  const float* tw1   = (const float*)d_in[9];
  const float* tb1   = (const float*)d_in[10];
  const float* gw0   = (const float*)d_in[11];
  const float* gb0   = (const float*)d_in[12];
  const float* gw1   = (const float*)d_in[13];
  const float* gb1   = (const float*)d_in[14];
  char*  ws  = (char*)d_ws;
  float* out = (float*)d_out;

  const int prepTotal = 6144 + 100352 + 53248 + 65536 + 409600;   // 634,880
  prep_kernel<<<(prepTotal + 255) / 256, 256, 0, stream>>>(
      cv, f2, f3, f4, f5, wproj, gw0, tw0, gw1, tw1, ws);
  conv_kernel<<<800, 256, 0, stream>>>(char_idxs, ws);
  head_kernel<<<200, 256, 0, stream>>>(ws, tb0, gb0, tb1, gb1, out);
}
